// GCN_50096498541255
// MI455X (gfx1250) — compile-verified, matrix operand order checked by reading, not run on hardware
//
#include <hip/hip_runtime.h>

#define F_IN 512
#define HID  16
#define NCLS 40

typedef __attribute__((ext_vector_type(16))) __bf16 v16bf;
typedef __attribute__((ext_vector_type(8)))  float  v8f;

// ---------------------------------------------------------------- utilities
__global__ void k_zero(float* __restrict__ p, long n) {
  long i = (long)blockIdx.x * blockDim.x + threadIdx.x;
  long s = (long)gridDim.x * blockDim.x;
  for (; i < n; i += s) p[i] = 0.0f;
}

__global__ void k_deg(const int* __restrict__ src, const int* __restrict__ dst,
                      float* __restrict__ odeg, float* __restrict__ ideg, int E) {
  int e = blockIdx.x * blockDim.x + threadIdx.x;
  if (e < E) {
    atomicAdd(&odeg[src[e]], 1.0f);
    atomicAdd(&ideg[dst[e]], 1.0f);
  }
}

__global__ void k_norm(float* __restrict__ p, int n) {  // deg -> clip(deg,1)^-0.5
  int i = blockIdx.x * blockDim.x + threadIdx.x;
  if (i < n) {
    float d = p[i];
    p[i] = rsqrtf(d < 1.0f ? 1.0f : d);
  }
}

// ------------------------------------------------- W1 repack into B-operand
// WMMA 16x16x32 bf16 B layout: lane L holds column n=L&15; element i holds
// k = kk*32 + ((i>>3)<<4) + ((L>>4)<<3) + (i&7).  Packed contiguously so the
// GEMM wave loads 32B per lane per K-step.
__global__ void k_packW1(const float* __restrict__ W1, __bf16* __restrict__ pack) {
  int t = blockIdx.x * blockDim.x + threadIdx.x;   // 16 kk * 32 lanes * 16 elems
  if (t >= 16 * 32 * 16) return;
  int i    = t & 15;
  int lane = (t >> 4) & 31;
  int kk   = t >> 9;
  int n = lane & 15;
  int k = kk * 32 + ((i >> 3) << 4) + ((lane >> 4) << 3) + (i & 7);
  pack[t] = (__bf16)W1[k * HID + n];
}

// -------------------------------------------------- layer-1 dense transform
// xw1[n,:] = (feats[n,:] * norm_src[n]) @ W1, via v_wmma_f32_16x16x32_bf16.
// One wave per 16-node tile, 8 waves per block.  Per kk step a wave touches
// exactly one 128B cacheline of each of its 16 feature rows.
__global__ void k_gemm1(const float* __restrict__ feats,
                        const float* __restrict__ nsrc,
                        const __bf16* __restrict__ w1p,
                        float* __restrict__ xw1, int N) {
  int lane = threadIdx.x & 31;
  int wave = threadIdx.x >> 5;
  int base = (blockIdx.x * 8 + wave) * 16;
  if (base >= N) return;                       // uniform per wave: EXEC stays full

  int row    = base + (lane & 15);
  int rclamp = row < N ? row : N - 1;          // clamp, mask on store
  float scale = nsrc[rclamp];
  const float* fp = feats + (long)rclamp * F_IN + ((lane >> 4) << 3);

  v8f acc = {};
  for (int kk = 0; kk < F_IN / 32; ++kk) {
    const float4* p0 = (const float4*)(fp + kk * 32);
    const float4* p1 = (const float4*)(fp + kk * 32 + 16);
    float4 f0 = p0[0], f1 = p0[1];
    float4 f2 = p1[0], f3 = p1[1];
    v16bf a;
    a[0]  = (__bf16)(f0.x * scale); a[1]  = (__bf16)(f0.y * scale);
    a[2]  = (__bf16)(f0.z * scale); a[3]  = (__bf16)(f0.w * scale);
    a[4]  = (__bf16)(f1.x * scale); a[5]  = (__bf16)(f1.y * scale);
    a[6]  = (__bf16)(f1.z * scale); a[7]  = (__bf16)(f1.w * scale);
    a[8]  = (__bf16)(f2.x * scale); a[9]  = (__bf16)(f2.y * scale);
    a[10] = (__bf16)(f2.z * scale); a[11] = (__bf16)(f2.w * scale);
    a[12] = (__bf16)(f3.x * scale); a[13] = (__bf16)(f3.y * scale);
    a[14] = (__bf16)(f3.z * scale); a[15] = (__bf16)(f3.w * scale);
    v16bf b = *(const v16bf*)(w1p + ((kk * 32 + lane) << 4));
    acc = __builtin_amdgcn_wmma_f32_16x16x32_bf16(false, a, false, b,
                                                  (short)0, acc, false, false);
  }

  // C/D layout: VGPR v -> row base + v + 8*(lane>>4), column lane&15.
  // Uniform guard: a wave's tile is all-in or ragged; common path is 8
  // unconditional b32 stores off one base address (immediate offsets).
  int col   = lane & 15;
  int rbase = base + ((lane >> 4) << 3);
  float* outp = xw1 + (long)rbase * HID + col;
  if (base + 16 <= N) {
#pragma unroll
    for (int v = 0; v < 8; ++v) outp[v * HID] = acc[v];
  } else {
#pragma unroll
    for (int v = 0; v < 8; ++v)
      if (rbase + v < N) outp[v * HID] = acc[v];
  }
}

// ----------------------------------------------------- layer-1 edge scatter
__global__ void k_scatter16(const float* __restrict__ xw1,
                            const int* __restrict__ src, const int* __restrict__ dst,
                            float* __restrict__ agg, int E) {
  long t = (long)blockIdx.x * blockDim.x + threadIdx.x;
  if (t >= (long)E * HID) return;
  int e = (int)(t >> 4);
  int f = (int)(t & 15);
  float v = xw1[(long)src[e] * HID + f];
  atomicAdd(&agg[(long)dst[e] * HID + f], v);
}

// h = relu(agg*norm_dst + b1) (in place in d_out); xd = h*dropout*norm_src
__global__ void k_finish1(float* __restrict__ h,
                          const float* __restrict__ ndst, const float* __restrict__ nsrc,
                          const float* __restrict__ b1, const float* __restrict__ dmask,
                          float* __restrict__ xd, int N) {
  long t = (long)blockIdx.x * blockDim.x + threadIdx.x;
  if (t >= (long)N * HID) return;
  int n = (int)(t >> 4);
  int f = (int)(t & 15);
  float v = h[t] * ndst[n] + b1[f];
  v = v > 0.0f ? v : 0.0f;
  h[t]  = v;
  xd[t] = v * dmask[t] * nsrc[n];
}

// ------------------------------------------- layer-2 tiny transform (K=16)
__global__ void k_gemm2(const float* __restrict__ xd, const float* __restrict__ W2,
                        float* __restrict__ xw2, int N) {
  long t = (long)blockIdx.x * blockDim.x + threadIdx.x;
  if (t >= (long)N * NCLS) return;
  int n = (int)(t / NCLS);
  int c = (int)(t % NCLS);
  const float* x = xd + (long)n * HID;
  float s = 0.0f;
#pragma unroll
  for (int k = 0; k < HID; ++k) s = fmaf(x[k], W2[k * NCLS + c], s);
  xw2[t] = s;
}

__global__ void k_scatter40(const float* __restrict__ xw2,
                            const int* __restrict__ src, const int* __restrict__ dst,
                            float* __restrict__ out2, int E) {
  long t = (long)blockIdx.x * blockDim.x + threadIdx.x;
  if (t >= (long)E * NCLS) return;
  int e = (int)(t / NCLS);
  int c = (int)(t % NCLS);
  float v = xw2[(long)src[e] * NCLS + c];
  atomicAdd(&out2[(long)dst[e] * NCLS + c], v);
}

// out = log_softmax(agg*norm_dst + b2) in place
__global__ void k_finish2(float* __restrict__ out2, const float* __restrict__ ndst,
                          const float* __restrict__ b2, int N) {
  int n = blockIdx.x * blockDim.x + threadIdx.x;
  if (n >= N) return;
  float nd = ndst[n];
  float v[NCLS];
  float m = -3.4e38f;
#pragma unroll
  for (int c = 0; c < NCLS; ++c) {
    v[c] = out2[(long)n * NCLS + c] * nd + b2[c];
    m = fmaxf(m, v[c]);
  }
  float s = 0.0f;
#pragma unroll
  for (int c = 0; c < NCLS; ++c) s += __expf(v[c] - m);
  float ls = __logf(s) + m;
#pragma unroll
  for (int c = 0; c < NCLS; ++c) out2[(long)n * NCLS + c] = v[c] - ls;
}

// ---------------------------------------------------------------- launcher
extern "C" void kernel_launch(void* const* d_in, const int* in_sizes, int n_in,
                              void* d_out, int out_size, void* d_ws, size_t ws_size,
                              hipStream_t stream) {
  const float* feats = (const float*)d_in[0];
  const float* W1    = (const float*)d_in[1];
  const float* b1    = (const float*)d_in[2];
  const float* W2    = (const float*)d_in[3];
  const float* b2    = (const float*)d_in[4];
  const float* dmask = (const float*)d_in[5];
  const int*   src   = (const int*)d_in[6];
  const int*   dst   = (const int*)d_in[7];
  const int N = in_sizes[0] / F_IN;
  const int E = in_sizes[6];

  float* out   = (float*)d_out;
  float* out_h = out;                    // N*16 : h, also layer-1 accumulator
  float* out_y = out + (long)N * HID;    // N*40 : log_softmax, also layer-2 acc

  float*  ws   = (float*)d_ws;
  float*  nsrc = ws;                     // N
  float*  ndst = ws + N;                 // N (contiguous with nsrc)
  float*  xw1  = ws + 2L * N;            // N*16
  float*  xd   = ws + 18L * N;           // N*16
  float*  xw2  = ws + 34L * N;           // N*40
  __bf16* w1p  = (__bf16*)(ws + 74L * N);// 8192 bf16

  const int B = 256;

  // zero accumulators (atomics accumulate; must reset every call)
  k_zero<<<2048, B, 0, stream>>>(out, (long)out_size);
  k_zero<<<1024, B, 0, stream>>>(nsrc, 2L * N);

  // degrees -> symmetric norms
  k_deg <<<(E + B - 1) / B, B, 0, stream>>>(src, dst, nsrc, ndst, E);
  k_norm<<<(2 * N + B - 1) / B, B, 0, stream>>>(nsrc, 2 * N);

  // layer 1: WMMA dense transform, then gather/scatter, then relu+dropout
  k_packW1<<<(16 * 32 * 16 + B - 1) / B, B, 0, stream>>>(W1, w1p);
  k_gemm1 <<<(N + 127) / 128, B, 0, stream>>>(feats, nsrc, w1p, xw1, N);
  k_scatter16<<<(int)(((long)E * HID + B - 1) / B), B, 0, stream>>>(xw1, src, dst, out_h, E);
  k_finish1<<<(int)(((long)N * HID + B - 1) / B), B, 0, stream>>>(out_h, ndst, nsrc, b1, dmask, xd, N);

  // layer 2: tiny transform, gather/scatter, log-softmax
  k_gemm2   <<<(int)(((long)N * NCLS + B - 1) / B), B, 0, stream>>>(xd, W2, xw2, N);
  k_scatter40<<<(int)(((long)E * NCLS + B - 1) / B), B, 0, stream>>>(xw2, src, dst, out_y, E);
  k_finish2 <<<(N + B - 1) / B, B, 0, stream>>>(out_y, ndst, b2, N);
}
